// StrnnLocPredictor_64218351010371
// MI455X (gfx1250) — compile-verified, matrix operand order checked by reading, not run on hardware
//
#include <hip/hip_runtime.h>
#include <hip/hip_bf16.h>

// ---------------- problem constants ----------------
#define BB 32
#define SS 128
#define DD 128
#define HH 128
#define EE 64
#define KK 11          // NUM_SLOTS + 1
#define KE 704         // KK * EE
#define PRE 16
#define OUTN 10000
#define H4 512         // 4*H

typedef __attribute__((ext_vector_type(16))) __bf16 bf16x16;
typedef __attribute__((ext_vector_type(8)))  float  f32x8;

// ---------------------------------------------------------------------------
// Fragment loader for V_WMMA_F32_16X16X32_BF16 (wave32).
// 16-bit A-matrix 16x32 layout (ISA 7.12.2):
//   lanes 0-15:  row M=lane,    VGPR0..3 = K 0..7,  VGPR4..7 = K 16..23
//   lanes 16-31: row M=lane-16, VGPR0..3 = K 8..15, VGPR4..7 = K 24..31
// With row-major data (ld elements/row) both halves are contiguous 16B loads.
// B uses the symmetric layout with Bt stored N-major (row n over k).
// ---------------------------------------------------------------------------
__device__ inline bf16x16 load_frag(const __bf16* base, int ld, int lane) {
  const __bf16* p = base + (size_t)(lane & 15) * ld + ((lane >> 4) << 3);
  bf16x16 f;
#pragma unroll
  for (int u = 0; u < 8; ++u) {
    f[u]     = p[u];       // K + 0..7   (one 16B load)
    f[u + 8] = p[u + 16];  // K + 16..23 (one 16B load)
  }
  return f;
}

__device__ inline f32x8 wmma_bf16(bf16x16 a, bf16x16 b, f32x8 c) {
  return __builtin_amdgcn_wmma_f32_16x16x32_bf16(false, a, false, b,
                                                 (short)0, c, false, false);
}

// fast activations: v_exp_f32 + raw v_rcp_f32 (no IEEE div fixup chain)
__device__ inline float fast_sigmoid(float z) {
  return __builtin_amdgcn_rcpf(1.0f + __expf(-z));
}
__device__ inline float fast_tanh(float x) {
  float e = __expf(2.0f * x);                       // inf-safe: rcp(inf)=0
  return 1.0f - 2.0f * __builtin_amdgcn_rcpf(e + 1.0f);
}

// ---------------------------------------------------------------------------
// Transpose f32 (R x C) -> bf16 (C x R), batched over blockIdx.z.
// ---------------------------------------------------------------------------
__global__ void __launch_bounds__(256)
k_transpose_bf16(const float* __restrict__ in, __bf16* __restrict__ out,
                 int R, int C) {
  __shared__ float tile[16][17];
  size_t boff = (size_t)blockIdx.z * R * C;
  in  += boff;
  out += boff;
  int r0 = blockIdx.y * 16, c0 = blockIdx.x * 16;
  int tx = threadIdx.x, ty = threadIdx.y;
  int r = r0 + ty, c = c0 + tx;
  if (r < R && c < C) tile[ty][tx] = in[(size_t)r * C + c];
  __syncthreads();
  int orow = c0 + ty, ocol = r0 + tx;
  if (orow < C && ocol < R) out[(size_t)orow * R + ocol] = (__bf16)tile[tx][ty];
}

// ---------------------------------------------------------------------------
// x = embed_table[full_seq]  ->  bf16 (B*S, D)
// ---------------------------------------------------------------------------
__global__ void __launch_bounds__(256)
k_embed(const int* __restrict__ seq, const float* __restrict__ table,
        __bf16* __restrict__ x) {
  int t = blockIdx.x * blockDim.x + threadIdx.x;
  if (t >= BB * SS * DD) return;
  int row = t >> 7;          // b*S + s
  int d   = t & (DD - 1);
  x[t] = (__bf16)table[(size_t)seq[row] * DD + d];
}

// ---------------------------------------------------------------------------
// bf16 WMMA GEMM: C(MxN) = A(MxK) * Bt(NxK)^T.
// Each wave owns a 2x4 block of 16x16 tiles (32 rows x 64 cols): the A
// fragment pair is reused across 4 N-tiles, the B fragment across 2 M-tiles
// -> 8 WMMAs per 6 fragment loads per k-step. N-tail guarded wave-uniformly
// (EXEC stays all-ones around WMMA). M must be a multiple of 32.
// EPI: 0 = store f32, 1 = store bf16, 2 = bf16(tanh(acc+bias)), 3 = f32 acc+bias
// ---------------------------------------------------------------------------
template <int EPI>
__global__ void __launch_bounds__(256)
k_gemm(const __bf16* __restrict__ A, const __bf16* __restrict__ Bt,
       const float* __restrict__ bias, void* __restrict__ Cout,
       int N, int Kd, int ldc) {
  int lane   = threadIdx.x & 31;
  int wave   = threadIdx.x >> 5;
  int wtile  = blockIdx.x * (blockDim.x >> 5) + wave;  // group of 4 ntiles
  int ntile0 = wtile * 4;
  int mtile0 = blockIdx.y * 2;
  if (ntile0 * 16 >= N) return;
  int ntc = (N / 16) - ntile0;
  if (ntc > 4) ntc = 4;                                // wave-uniform

  const __bf16* Ab0 = A + (size_t)(mtile0 + 0) * 16 * Kd;
  const __bf16* Ab1 = A + (size_t)(mtile0 + 1) * 16 * Kd;

  f32x8 acc[2][4] = {};
  for (int k0 = 0; k0 < Kd; k0 += 32) {
    if (k0 + 32 < Kd) {                                // L2 prefetch next chunk
      __builtin_prefetch(Ab0 + k0 + 32, 0, 0);
      __builtin_prefetch(Bt + (size_t)ntile0 * 16 * Kd + k0 + 32, 0, 0);
    }
    bf16x16 a0 = load_frag(Ab0 + k0, Kd, lane);
    bf16x16 a1 = load_frag(Ab1 + k0, Kd, lane);
#pragma unroll
    for (int t = 0; t < 4; ++t) {
      if (t < ntc) {
        bf16x16 bf = load_frag(Bt + (size_t)(ntile0 + t) * 16 * Kd + k0, Kd, lane);
        acc[0][t] = wmma_bf16(a0, bf, acc[0][t]);
        acc[1][t] = wmma_bf16(a1, bf, acc[1][t]);
      }
    }
  }

  int rb = ((lane >> 4) << 3);
#pragma unroll
  for (int t = 0; t < 4; ++t) {
    if (t >= ntc) break;
    int col = (ntile0 + t) * 16 + (lane & 15);
    float bv = (EPI == 2 || EPI == 3) ? bias[col] : 0.0f;
#pragma unroll
    for (int mm = 0; mm < 2; ++mm) {
      int rbase = (mtile0 + mm) * 16 + rb;
#pragma unroll
      for (int v = 0; v < 8; ++v) {
        size_t idx = (size_t)(rbase + v) * ldc + col;
        float a = acc[mm][t][v];
        if (EPI == 0)      ((float*)Cout)[idx]  = a;
        else if (EPI == 1) ((__bf16*)Cout)[idx] = (__bf16)a;
        else if (EPI == 2) ((__bf16*)Cout)[idx] = (__bf16)fast_tanh(a + bv);
        else               ((float*)Cout)[idx]  = a + bv;
      }
    }
  }
}

// ---------------------------------------------------------------------------
// Spatiotemporal aggregation. One wave per (b,i):
//   agg[b,i,dc,e] = sum_{j<=i, mask} v_slots[b,j,tc(b,i,j),e]
// Lane L owns e = {L, L+32}. dc-select via predicated adds (no scratch).
// ---------------------------------------------------------------------------
__global__ void __launch_bounds__(256)
k_aggregate(const float* __restrict__ ts, const float* __restrict__ lat,
            const float* __restrict__ lng, const int* __restrict__ valid_len,
            const __bf16* __restrict__ v_slots, __bf16* __restrict__ agg) {
  int gwave = (blockIdx.x * blockDim.x + threadIdx.x) >> 5;
  int lane  = threadIdx.x & 31;
  if (gwave >= BB * SS) return;
  int b = gwave >> 7;
  int i = gwave & (SS - 1);

  float ti   = ts[b * SS + i];
  float lati = lat[b * SS + i];
  float lngi = lng[b * SS + i];
  int   vl   = valid_len[b];

  float acc0[KK], acc1[KK];
#pragma unroll
  for (int k = 0; k < KK; ++k) { acc0[k] = 0.0f; acc1[k] = 0.0f; }

  int jmax = (i < vl - 1) ? i : (vl - 1);
  for (int j = 0; j <= jmax; ++j) {
    float td = ti - ts[b * SS + j];
    if (td < 0.0f || td > 3600.0f) continue;
    float dx = lati - lat[b * SS + j];
    float dy = lngi - lng[b * SS + j];
    float dist = sqrtf(dx * dx + dy * dy);
    int tc = (int)floorf(td * (10.0f / 3600.0f));    if (tc > 10) tc = 10;
    int dc = (int)floorf(fminf(dist, 1.0f) * 10.0f); if (dc > 10) dc = 10;

    const __bf16* vp = v_slots + ((size_t)(b * SS + j) * KE + tc * EE);
    float v0 = (float)vp[lane];
    float v1 = (float)vp[lane + 32];
#pragma unroll
    for (int k = 0; k < KK; ++k) {
      acc0[k] += (k == dc) ? v0 : 0.0f;   // v_cndmask + v_add
      acc1[k] += (k == dc) ? v1 : 0.0f;
    }
  }

  __bf16* ap = agg + (size_t)gwave * KE;
#pragma unroll
  for (int k = 0; k < KK; ++k) {
    ap[k * EE + lane]      = (__bf16)acc0[k];
    ap[k * EE + lane + 32] = (__bf16)acc1[k];
  }
}

// ---------------------------------------------------------------------------
// RNN scan: h_new = sigmoid(x_cand[:,i,:] + h @ hidden_weights), 128 steps.
// Single workgroup (16 waves = 512 thr) on one WGP; h lives in LDS as bf16;
// the 8 N-tiles of HWt stay resident in registers (4 k-step fragments/wave).
// Steps S-PRE-1 .. S-2 are emitted as the MLP's A matrix (row = b*PRE + t).
// ---------------------------------------------------------------------------
__global__ void __launch_bounds__(512)
k_rnn(const float* __restrict__ x_cand, const __bf16* __restrict__ HWt,
      __bf16* __restrict__ rnn_pre) {
  __shared__ __bf16 hbuf[BB * HH];

  int tid  = threadIdx.x;
  int lane = tid & 31;
  int wave = tid >> 5;        // 0..15
  int mt   = wave >> 3;       // 0..1  (M tile of 32 batch rows)
  int nt   = wave & 7;        // 0..7  (N tile of 128 hidden units)

  bf16x16 Bfrag[4];
#pragma unroll
  for (int kk = 0; kk < 4; ++kk)
    Bfrag[kk] = load_frag(HWt + (size_t)nt * 16 * HH + kk * 32, HH, lane);

  for (int t = tid; t < BB * HH; t += blockDim.x) hbuf[t] = (__bf16)0.0f;
  __syncthreads();

  int col   = nt * 16 + (lane & 15);
  int rbase = mt * 16 + ((lane >> 4) << 3);

  for (int i = 0; i < SS; ++i) {
    f32x8 acc = {};
#pragma unroll
    for (int kk = 0; kk < 4; ++kk) {
      bf16x16 af = load_frag(hbuf + mt * 16 * HH + kk * 32, HH, lane);
      acc = wmma_bf16(af, Bfrag[kk], acc);
    }

    float hn[8];
#pragma unroll
    for (int v = 0; v < 8; ++v) {
      int brow = rbase + v;
      float z = acc[v] + x_cand[((size_t)brow * SS + i) * HH + col];
      hn[v] = fast_sigmoid(z);          // v_exp + v_rcp, no IEEE div chain
    }

    __syncthreads();   // all A reads of hbuf done before overwrite
#pragma unroll
    for (int v = 0; v < 8; ++v)
      hbuf[(rbase + v) * HH + col] = (__bf16)hn[v];

    if (i >= SS - PRE - 1 && i < SS - 1) {
      int t2 = i - (SS - PRE - 1);
#pragma unroll
      for (int v = 0; v < 8; ++v)
        rnn_pre[((size_t)(rbase + v) * PRE + t2) * HH + col] = (__bf16)hn[v];
    }
    __syncthreads();
  }
}

// ---------------------------------------------------------------------------
extern "C" void kernel_launch(void* const* d_in, const int* in_sizes, int n_in,
                              void* d_out, int out_size, void* d_ws,
                              size_t ws_size, hipStream_t stream) {
  const int*   full_seq       = (const int*)d_in[0];
  const int*   valid_len      = (const int*)d_in[1];
  /* d_in[2] = pre_len (compile-time PRE) */
  const float* timestamp      = (const float*)d_in[3];
  const float* lat            = (const float*)d_in[4];
  const float* lng            = (const float*)d_in[5];
  const float* embed_table    = (const float*)d_in[6];
  const float* time_weights   = (const float*)d_in[7];
  const float* dist_weights   = (const float*)d_in[8];
  const float* hidden_weights = (const float*)d_in[9];
  const float* W1             = (const float*)d_in[10];
  const float* b1             = (const float*)d_in[11];
  const float* W2             = (const float*)d_in[12];
  const float* b2             = (const float*)d_in[13];

  char* ws = (char*)d_ws;
  size_t off = 0;
  auto alloc = [&](size_t bytes) -> void* {
    void* p = ws + off;
    off = (off + bytes + 255) & ~(size_t)255;
    return p;
  };

  __bf16* TWt     = (__bf16*)alloc((size_t)KK * EE * DD * 2);   // (k, E, D)
  __bf16* DWt     = (__bf16*)alloc((size_t)HH * KE * 2);        // (H, KE)
  __bf16* HWt     = (__bf16*)alloc((size_t)HH * HH * 2);        // (Hout, Hin)
  __bf16* W1t     = (__bf16*)alloc((size_t)H4 * HH * 2);        // (4H, H)
  __bf16* W2t     = (__bf16*)alloc((size_t)OUTN * H4 * 2);      // (OUT, 4H)
  __bf16* x_bf    = (__bf16*)alloc((size_t)BB * SS * DD * 2);
  __bf16* v_slots = (__bf16*)alloc((size_t)BB * SS * KE * 2);
  __bf16* agg     = (__bf16*)alloc((size_t)BB * SS * KE * 2);
  float*  x_cand  = (float*) alloc((size_t)BB * SS * HH * 4);
  __bf16* rnn_pre = (__bf16*)alloc((size_t)BB * PRE * HH * 2);
  __bf16* hidden  = (__bf16*)alloc((size_t)BB * PRE * H4 * 2);
  (void)ws_size; (void)in_sizes; (void)n_in; (void)out_size;

  dim3 tb(16, 16);
  // weight prep: f32 -> bf16, N-major for contiguous B fragments
  k_transpose_bf16<<<dim3(4, 8, KK), tb, 0, stream>>>(time_weights, TWt, DD, EE);
  k_transpose_bf16<<<dim3(8, 44, 1), tb, 0, stream>>>(dist_weights, DWt, KE, HH);
  k_transpose_bf16<<<dim3(8, 8, 1),  tb, 0, stream>>>(hidden_weights, HWt, HH, HH);
  k_transpose_bf16<<<dim3(32, 8, 1), tb, 0, stream>>>(W1, W1t, HH, H4);
  k_transpose_bf16<<<dim3(625, 32, 1), tb, 0, stream>>>(W2, W2t, H4, OUTN);

  // embedding gather -> bf16
  k_embed<<<(BB * SS * DD + 255) / 256, 256, 0, stream>>>(full_seq, embed_table, x_bf);

  // v_slots[:, k*64 .. k*64+63] = x @ time_weights[k]
  // M=4096, N=64 (4 ntiles = 1 wave-group), K=128
  for (int k = 0; k < KK; ++k)
    k_gemm<1><<<dim3(1, BB * SS / 32), 32, 0, stream>>>(
        x_bf, TWt + (size_t)k * EE * DD, nullptr,
        v_slots + (size_t)k * EE, EE, DD, KE);

  // masked gather/scatter aggregation (one wave per (b,i))
  k_aggregate<<<(BB * SS * 32 + 255) / 256, 256, 0, stream>>>(
      timestamp, lat, lng, valid_len, v_slots, agg);

  // x_cand = agg @ dist_weights   (M=4096, N=128 -> 2 wave-groups, K=704)
  k_gemm<0><<<dim3(1, BB * SS / 32), 64, 0, stream>>>(
      agg, DWt, nullptr, x_cand, HH, KE, HH);

  // sequential RNN, single workgroup, WMMA per step
  k_rnn<<<1, 512, 0, stream>>>(x_cand, HWt, rnn_pre);

  // hidden = tanh(rnn_pre @ W1 + b1)   (M=512, N=512 -> 8 wave-groups, K=128)
  k_gemm<2><<<dim3(1, BB * PRE / 32), 256, 0, stream>>>(
      rnn_pre, W1t, b1, hidden, H4, HH, H4);

  // out = hidden @ W2 + b2   (M=512, N=10000, K=512) — FLOP-dominant stage
  // 625 ntiles -> 157 wave-groups -> 20 blocks of 8 waves
  k_gemm<3><<<dim3((OUTN / 16 + 31) / 32, BB * PRE / 32), 256, 0, stream>>>(
      hidden, W2t, b2, d_out, OUTN, H4, OUTN);
}
